// CrossAttention_12197707120712
// MI455X (gfx1250) — matrix operand order repaired and checked by reading, not compiled, not run on hardware
//
#include <hip/hip_runtime.h>
#include <hip/hip_bf16.h>

typedef _Float16 half_t;
typedef __attribute__((ext_vector_type(16))) _Float16 v16h;
typedef __attribute__((ext_vector_type(8)))  _Float16 v8h;
typedef __attribute__((ext_vector_type(8)))  float    v8f;
typedef __attribute__((ext_vector_type(4)))  unsigned int u32x4;
typedef __attribute__((ext_vector_type(8)))  int      i32x8;
typedef __attribute__((ext_vector_type(4)))  int      i32x4;

#define DIM 512            // model dim == inner dim (8 heads * 64)
#define NSEQ 512           // sequence length per page
#define NPAGES 32          // b*p = 4*8
#define NROWS (NPAGES*NSEQ)
#define DHEAD 64
#define ATTN_SCALE 0.125f  // 64^-0.5

static __device__ __forceinline__ v8f wmma16(v16h a, v16h b, v8f c) {
    return __builtin_amdgcn_wmma_f32_16x16x32_f16(
        false, a, false, b, (short)0, c, false, false);
}
static __device__ __forceinline__ v16h cat8(v8h lo, v8h hi) {
    return __builtin_shufflevector(lo, hi, 0,1,2,3,4,5,6,7,8,9,10,11,12,13,14,15);
}

// ---------------------------------------------------------------------------
// Elementwise fp32 -> f16 (optionally scaled). n multiple of 2048.
// ---------------------------------------------------------------------------
__global__ __launch_bounds__(256) void cvt_f32_f16(
    const float* __restrict__ src, half_t* __restrict__ dst, int n, float scale)
{
    int i = (blockIdx.x * 256 + threadIdx.x) * 8;
    if (i + 8 <= n) {
        float4 a = *(const float4*)(src + i);
        float4 b = *(const float4*)(src + i + 4);
        v8h o;
        o[0] = (half_t)(a.x * scale); o[1] = (half_t)(a.y * scale);
        o[2] = (half_t)(a.z * scale); o[3] = (half_t)(a.w * scale);
        o[4] = (half_t)(b.x * scale); o[5] = (half_t)(b.y * scale);
        o[6] = (half_t)(b.z * scale); o[7] = (half_t)(b.w * scale);
        *(v8h*)(dst + i) = o;
    }
}

// ---------------------------------------------------------------------------
// TDM: async-load a 128-row x 32-half tile of a [16384x512] f16 tensor to LDS.
// D# per CDNA5 ISA ch.8: group0 = {count/lds_addr/global_addr/type},
// group1 = {data_size=2B, tensor_dim0=512, tensor_dim1=16384, tile 32x128,
// dim0_stride=512}.  clang-23 toolchain: 6-arg builtin.
// ---------------------------------------------------------------------------
static __device__ __forceinline__ void tdm_load_128x32(
    const half_t* gsrc, half_t* lds_dst)
{
    unsigned long long ga = (unsigned long long)(const void*)gsrc;
    unsigned int laddr   = (unsigned int)(unsigned long long)(void*)lds_dst;
    u32x4 g0;
    g0[0] = 1u;                                   // count=1, user descriptor
    g0[1] = laddr;                                // lds_addr
    g0[2] = (unsigned int)ga;                     // global_addr[31:0]
    g0[3] = (unsigned int)((ga >> 32) & 0x01FFFFFFull) | 0x80000000u; // addr[56:32] | type=2
    i32x8 g1;
    g1[0] = 0x00010000;                           // data_size = 2 bytes
    g1[1] = (int)(512u << 16);                    // tensor_dim0 = 512 (bits 79:48)
    g1[2] = (int)(16384u << 16);                  // tensor_dim1 = 16384 (bits 111:80)
    g1[3] = (int)(32u << 16);                     // tile_dim0 = 32 (bits 127:112)
    g1[4] = 128;                                  // tile_dim1 = 128
    g1[5] = 512;                                  // tensor_dim0_stride = 512
    g1[6] = 0;
    g1[7] = 0;
    i32x4 z4 = {0, 0, 0, 0};
    i32x8 z8 = {0, 0, 0, 0, 0, 0, 0, 0};
    __builtin_amdgcn_tensor_load_to_lds(g0, g1, z4, z4, z8, 0);
}

// ---------------------------------------------------------------------------
// GEMM body shared by both projection kernels: C[128x64] tile,
// 8 waves, each wave 32x32 (4 WMMAs / k-step). A staged by TDM, B staged
// transposed ([col][k]) with batched loads.
// ---------------------------------------------------------------------------
#define GEMM_PROLOG()                                                         \
    __shared__ alignas(64) half_t As[128 * 32];                               \
    __shared__ alignas(64) half_t Bs[64 * 32];                                \
    const int t      = threadIdx.x;                                           \
    const int lane   = t & 31;                                                \
    const int w      = t >> 5;                                                \
    const int rowblk = w >> 1;                                                \
    const int colblk = w & 1;                                                 \
    const int lo     = lane & 15;                                             \
    const int hi     = lane >> 4;                                             \
    const int row0   = blockIdx.y * 128;                                      \
    const int col0   = blockIdx.x * 64;                                       \
    const int bcol   = t & 63;                                                \
    const int brow0  = (t >> 6) * 8;                                          \
    v8f acc[2][2] = {};                                                       \
    for (int kk = 0; kk < DIM; kk += 32) {                                    \
        if (w == 0) tdm_load_128x32(X + (size_t)row0 * DIM + kk, As);         \
        v8h bb;                                                               \
        _Pragma("unroll")                                                     \
        for (int i = 0; i < 8; ++i)                                           \
            bb[i] = W[(size_t)(kk + brow0 + i) * DIM + col0 + bcol];          \
        *(v8h*)(Bs + bcol * 32 + brow0) = bb;                                 \
        if (kk + 32 < DIM)                                                    \
            __builtin_prefetch(W + (size_t)(kk + 32 + brow0) * DIM + col0 + bcol, 0, 1); \
        __builtin_amdgcn_s_wait_tensorcnt(0);                                 \
        __syncthreads();                                                      \
        v16h a[2], b[2];                                                      \
        _Pragma("unroll")                                                     \
        for (int i = 0; i < 2; ++i) {                                         \
            const int mrow = rowblk * 32 + i * 16 + lo;                       \
            a[i] = cat8(*(const v8h*)(As + mrow * 32 + hi * 8),               \
                        *(const v8h*)(As + mrow * 32 + hi * 8 + 16));         \
            const int n0 = colblk * 32 + i * 16 + lo;                         \
            b[i] = cat8(*(const v8h*)(Bs + n0 * 32 + hi * 8),                 \
                        *(const v8h*)(Bs + n0 * 32 + hi * 8 + 16));           \
        }                                                                     \
        _Pragma("unroll")                                                     \
        for (int ar = 0; ar < 2; ++ar)                                        \
            _Pragma("unroll")                                                 \
            for (int bc = 0; bc < 2; ++bc)                                    \
                acc[ar][bc] = wmma16(a[ar], b[bc], acc[ar][bc]);              \
        __syncthreads();                                                      \
    }

// dst(f16) = X(f16)[16384x512] @ W(f16)[512x512]; vt_mode=0: per-head
// [bh][n][d]; vt_mode=1: per-head transposed [bh][d][n] (for V).
__global__ __launch_bounds__(256) void gemm_proj(
    const half_t* __restrict__ X, const half_t* __restrict__ W,
    half_t* __restrict__ dst, int vt_mode)
{
    GEMM_PROLOG()
    #pragma unroll
    for (int ar = 0; ar < 2; ++ar)
        #pragma unroll
        for (int bc = 0; bc < 2; ++bc) {
            int col = col0 + colblk * 32 + bc * 16 + lo;
            int h = col >> 6, d = col & 63;
            #pragma unroll
            for (int r = 0; r < 8; ++r) {
                int g = row0 + rowblk * 32 + ar * 16 + r + 8 * hi;
                int page = g >> 9, n = g & 511;
                size_t off = vt_mode
                    ? ((size_t)((page * 8 + h) * 64 + d) * NSEQ + n)
                    : ((size_t)((page * 8 + h) * NSEQ + n) * 64 + d);
                dst[off] = (half_t)acc[ar][bc][r];
            }
        }
}

// out(f32) = X(f16)[16384x512] @ W(f16)[512x512] + bias
__global__ __launch_bounds__(256) void gemm_out(
    const half_t* __restrict__ X, const half_t* __restrict__ W,
    const float* __restrict__ bias, float* __restrict__ out)
{
    GEMM_PROLOG()
    #pragma unroll
    for (int ar = 0; ar < 2; ++ar)
        #pragma unroll
        for (int bc = 0; bc < 2; ++bc) {
            int col = col0 + colblk * 32 + bc * 16 + lo;
            float bv = bias[col];
            #pragma unroll
            for (int r = 0; r < 8; ++r) {
                int g = row0 + rowblk * 32 + ar * 16 + r + 8 * hi;
                out[(size_t)g * DIM + col] = acc[ar][bc][r] + bv;
            }
        }
}

// ---------------------------------------------------------------------------
// Flash attention: 1 wave / block, 32 query rows (2 tiles) per wave so each
// loaded K/V fragment feeds 2 WMMAs. qh/kh: [bh][512][64] (Q pre-scaled via
// Wq), vt: [bh][64][512], oh out: [16384][512] f16.
// ---------------------------------------------------------------------------
__global__ __launch_bounds__(32) void attn(
    const half_t* __restrict__ qh, const half_t* __restrict__ kh,
    const half_t* __restrict__ vt, half_t* __restrict__ oh)
{
    __shared__ alignas(32) half_t Pl[2][16 * 32];

    const int lane = threadIdx.x;
    const int lo   = lane & 15;
    const int hi   = lane >> 4;
    const int bh   = blockIdx.y;          // page*8 + h
    const int q0   = blockIdx.x * 32;

    const half_t* Q = qh + (size_t)bh * NSEQ * DHEAD;
    const half_t* K = kh + (size_t)bh * NSEQ * DHEAD;
    const half_t* V = vt + (size_t)bh * DHEAD * NSEQ;

    v16h aq[2][2];
    #pragma unroll
    for (int qi = 0; qi < 2; ++qi)
        #pragma unroll
        for (int kc = 0; kc < 2; ++kc) {
            const half_t* base = Q + (size_t)(q0 + qi * 16 + lo) * DHEAD + kc * 32 + hi * 8;
            aq[qi][kc] = cat8(*(const v8h*)base, *(const v8h*)(base + 16));
        }

    float rmax[2][8], rsum[2][8];
    #pragma unroll
    for (int qi = 0; qi < 2; ++qi)
        #pragma unroll
        for (int r = 0; r < 8; ++r) { rmax[qi][r] = -3.0e38f; rsum[qi][r] = 0.f; }
    v8f oacc[2][4] = {};

    for (int j0 = 0; j0 < NSEQ; j0 += 32) {
        v8f s[2][2] = {};
        #pragma unroll
        for (int kc = 0; kc < 2; ++kc) {
            const half_t* kb0 = K + (size_t)(j0 + lo) * DHEAD + kc * 32 + hi * 8;
            const half_t* kb1 = K + (size_t)(j0 + 16 + lo) * DHEAD + kc * 32 + hi * 8;
            v16h b0 = cat8(*(const v8h*)kb0, *(const v8h*)(kb0 + 16));
            v16h b1 = cat8(*(const v8h*)kb1, *(const v8h*)(kb1 + 16));
            #pragma unroll
            for (int qi = 0; qi < 2; ++qi) {
                s[qi][0] = wmma16(aq[qi][kc], b0, s[qi][0]);
                s[qi][1] = wmma16(aq[qi][kc], b1, s[qi][1]);
            }
        }

        #pragma unroll
        for (int qi = 0; qi < 2; ++qi) {
            float corr[8];
            #pragma unroll
            for (int r = 0; r < 8; ++r) {
                float m = fmaxf(s[qi][0][r], s[qi][1][r]);
                m = fmaxf(m, __shfl_xor(m, 1, 32));
                m = fmaxf(m, __shfl_xor(m, 2, 32));
                m = fmaxf(m, __shfl_xor(m, 4, 32));
                m = fmaxf(m, __shfl_xor(m, 8, 32));
                float mn = fmaxf(rmax[qi][r], m);
                corr[r]  = __expf(rmax[qi][r] - mn);
                rmax[qi][r] = mn;
                float p0 = __expf(s[qi][0][r] - mn);
                float p1 = __expf(s[qi][1][r] - mn);
                float rs = p0 + p1;
                rs += __shfl_xor(rs, 1, 32);
                rs += __shfl_xor(rs, 2, 32);
                rs += __shfl_xor(rs, 4, 32);
                rs += __shfl_xor(rs, 8, 32);
                rsum[qi][r] = rsum[qi][r] * corr[r] + rs;
                Pl[qi][(r + 8 * hi) * 32 + lo]      = (half_t)p0;
                Pl[qi][(r + 8 * hi) * 32 + 16 + lo] = (half_t)p1;
            }
            #pragma unroll
            for (int tt = 0; tt < 4; ++tt)
                #pragma unroll
                for (int r = 0; r < 8; ++r) oacc[qi][tt][r] *= corr[r];
        }

        asm volatile("s_wait_dscnt 0" ::: "memory");

        v16h pa[2];
        #pragma unroll
        for (int qi = 0; qi < 2; ++qi)
            pa[qi] = cat8(*(const v8h*)(&Pl[qi][0] + lo * 32 + hi * 8),
                          *(const v8h*)(&Pl[qi][0] + lo * 32 + hi * 8 + 16));

        #pragma unroll
        for (int tt = 0; tt < 4; ++tt) {
            const half_t* vb = V + (size_t)(tt * 16 + lo) * NSEQ + j0 + hi * 8;
            v16h bv = cat8(*(const v8h*)vb, *(const v8h*)(vb + 16));
            #pragma unroll
            for (int qi = 0; qi < 2; ++qi)
                oacc[qi][tt] = wmma16(pa[qi], bv, oacc[qi][tt]);
        }
    }

    const int page = bh >> 3, h = bh & 7;
    #pragma unroll
    for (int qi = 0; qi < 2; ++qi)
        #pragma unroll
        for (int tt = 0; tt < 4; ++tt)
            #pragma unroll
            for (int r = 0; r < 8; ++r) {
                int n   = q0 + qi * 16 + r + 8 * hi;
                int col = h * 64 + tt * 16 + lo;
                oh[((size_t)page * NSEQ + n) * DIM + col] =
                    (half_t)(oacc[qi][tt][r] / rsum[qi][r]);
            }
}

// ---------------------------------------------------------------------------
extern "C" void kernel_launch(void* const* d_in, const int* in_sizes, int n_in,
                              void* d_out, int out_size, void* d_ws, size_t ws_size,
                              hipStream_t stream) {
    const float* q_in  = (const float*)d_in[0];
    const float* kv_in = (const float*)d_in[1];
    const float* Wq    = (const float*)d_in[2];
    const float* Wk    = (const float*)d_in[3];
    const float* Wv    = (const float*)d_in[4];
    const float* Wo    = (const float*)d_in[5];
    const float* bo    = (const float*)d_in[6];
    float* out = (float*)d_out;

    const size_t actE = (size_t)NROWS * DIM;  // 8.4M
    const size_t wE   = (size_t)DIM * DIM;    // 262144
    half_t* q16  = (half_t*)d_ws;
    half_t* kv16 = q16  + actE;
    half_t* wq16 = kv16 + actE;
    half_t* wk16 = wq16 + wE;
    half_t* wv16 = wk16 + wE;
    half_t* wo16 = wv16 + wE;
    half_t* qhb  = wo16 + wE;
    half_t* khb  = qhb  + actE;
    half_t* vtb  = khb  + actE;
    half_t* ohb  = vtb  + actE;

    // fp32 -> f16 streaming converts (scale folded into Wq)
    cvt_f32_f16<<<actE / 2048, 256, 0, stream>>>(q_in,  q16,  (int)actE, 1.0f);
    cvt_f32_f16<<<actE / 2048, 256, 0, stream>>>(kv_in, kv16, (int)actE, 1.0f);
    cvt_f32_f16<<<wE   / 2048, 256, 0, stream>>>(Wq,    wq16, (int)wE,   ATTN_SCALE);
    cvt_f32_f16<<<wE   / 2048, 256, 0, stream>>>(Wk,    wk16, (int)wE,   1.0f);
    cvt_f32_f16<<<wE   / 2048, 256, 0, stream>>>(Wv,    wv16, (int)wE,   1.0f);
    cvt_f32_f16<<<wE   / 2048, 256, 0, stream>>>(Wo,    wo16, (int)wE,   1.0f);

    dim3 gGemm(DIM / 64, NROWS / 128);        // (8, 128)
    gemm_proj<<<gGemm, 256, 0, stream>>>(q16,  wq16, qhb, 0);
    gemm_proj<<<gGemm, 256, 0, stream>>>(kv16, wk16, khb, 0);
    gemm_proj<<<gGemm, 256, 0, stream>>>(kv16, wv16, vtb, 1);

    dim3 gAttn(NSEQ / 32, NPAGES * 8);        // (16, 256)
    attn<<<gAttn, 32, 0, stream>>>(qhb, khb, vtb, ohb);

    gemm_out<<<gGemm, 256, 0, stream>>>(ohb, wo16, bo, out);
}
